// OnnxSafeHTDemucs_66305705115872
// MI455X (gfx1250) — matrix operand order repaired and checked by reading, not compiled, not run on hardware
//
#include <hip/hip_runtime.h>
#include <math.h>

// ---------------------------------------------------------------------------
// Problem constants (from reference)
// ---------------------------------------------------------------------------
constexpr int BB    = 4;
constexpr int CC    = 2;
constexpr int SS    = 4;
constexpr int TLEN  = 441000;
constexpr int NFFT  = 4096;
constexpr int HOP   = 1024;
constexpr int NF    = 2048;     // freqs actually used (Nyquist row sliced off / zero-padded)
constexpr int LE    = 431;      // ceil(T/HOP)
constexpr int MB64  = 7;        // ceil(431/64) block-level M tiles
constexpr int NBC   = BB * CC;  // 8
constexpr int NBSC  = BB * SS * CC; // 32
constexpr float TWO_PI_OVER_N = 1.5339807878856412e-3f; // 2*pi/4096

// LDS staging geometry (per block): 2 buffers x 2 bases x 64 rows x 40 bf16
constexpr int ROW_BF   = 40;              // 80B row stride: 16B aligned, bank-conflict free
constexpr int BASIS_BF = 64 * ROW_BF;     // 2560
constexpr int PB_BF    = 2 * BASIS_BF;    // 5120 (one parity buffer, cos+sin)
constexpr int SMEM_BF  = 2 * PB_BF;       // 10240 bf16 = 20480 B

typedef __attribute__((ext_vector_type(16))) __bf16 v16bf;
typedef __attribute__((ext_vector_type(8)))  __bf16 v8bf;
typedef __attribute__((ext_vector_type(8)))  float  v8f;
typedef __attribute__((address_space(3)))    __bf16 lds_bf16;

static __device__ inline v8f wmma_bf16(v16bf a, v16bf b, v8f c) {
    return __builtin_amdgcn_wmma_f32_16x16x32_bf16(
        /*neg_a=*/false, a, /*neg_b=*/false, b,
        /*c_mod=*/(short)0, c, /*reuse_a=*/false, /*reuse_b=*/false);
}

// A-matrix per-lane fragment: 8 bf16 at p[0..7] (VGPR0-3) + 8 bf16 at p[8..15] (VGPR4-7)
static __device__ inline v16bf load_a16(const __bf16* p) {
    v8bf lo = *(const v8bf*)(p);
    v8bf hi = *(const v8bf*)(p + 8);
    return __builtin_shufflevector(lo, hi, 0,1,2,3,4,5,6,7,8,9,10,11,12,13,14,15);
}

// CDNA5 async DMA: 16B per lane, global -> LDS, tracked on ASYNCcnt
static __device__ inline void async_ld16(unsigned lds_addr, const void* gptr) {
    asm volatile("global_load_async_to_lds_b128 %0, %1, off"
                 :: "v"(lds_addr), "v"(gptr)
                 : "memory");
}
static __device__ inline void wait_async0() {
    asm volatile("s_wait_asynccnt 0x0" ::: "memory");
}

// ---------------------------------------------------------------------------
// K0: build DFT basis tables in bf16, both [f][t] (STFT-B) and [t][f] (iSTFT-B)
// ---------------------------------------------------------------------------
__global__ __launch_bounds__(256) void build_tables(
    __bf16* __restrict__ cosT, __bf16* __restrict__ sinT,
    __bf16* __restrict__ cosTT, __bf16* __restrict__ sinTT)
{
    int idx = blockIdx.x * 256 + threadIdx.x;          // < 2048*4096
    int f = idx >> 12;
    int t = idx & 4095;
    int m = (f * t) & 4095;                            // exact phase reduction
    float s, c;
    __sincosf((float)m * TWO_PI_OVER_N, &s, &c);
    cosT[idx] = (__bf16)c;
    sinT[idx] = (__bf16)s;
    size_t o2 = (size_t)t * NF + f;
    cosTT[o2] = (__bf16)c;
    sinTT[o2] = (__bf16)s;
}

// ---------------------------------------------------------------------------
// K1: windowed frames (bf16), composed reflect-pad folded into one index map.
// ---------------------------------------------------------------------------
__global__ __launch_bounds__(256) void build_frames(
    const float* __restrict__ mix, __bf16* __restrict__ frames)
{
    int idx = blockIdx.x * 256 + threadIdx.x;          // < 8*431*4096
    int t  = idx & 4095;
    int le = (idx >> 12) % LE;
    int bc = idx / (LE * NFFT);
    int r = (le + 2) * HOP + t - 2048 - 1536;          // in [-1536, 442879]
    if (r < 0)      r = -r;
    if (r >= TLEN)  r = 2 * TLEN - 2 - r;
    float w = 0.5f - 0.5f * __cosf((float)t * TWO_PI_OVER_N);
    frames[idx] = (__bf16)(mix[(size_t)bc * TLEN + r] * w);
}

// ---------------------------------------------------------------------------
// K2: STFT GEMM.  M=le(431) N=f(2048) K=t(4096).
// Block = 4 waves = 64(M)x64(N); cos/sin B tiles async-staged to LDS, double
// buffered; each wave computes a 16x64 strip for both bases.
// Inner step: burst-load all 8 B fragments from LDS, then 8 back-to-back WMMAs.
// ---------------------------------------------------------------------------
__global__ __launch_bounds__(128) void gemm_stft(
    const __bf16* __restrict__ frames,
    const __bf16* __restrict__ cosT, const __bf16* __restrict__ sinT,
    float* __restrict__ Zr, float* __restrict__ Si)
{
    __shared__ alignas(16) __bf16 smem[SMEM_BF];

    int lid   = threadIdx.x;           // 0..127
    int lane  = lid & 31;
    int wave  = lid >> 5;
    int bid   = blockIdx.x;            // < 8 * 7 * 32
    int bc    = bid / (MB64 * 32);
    int rem   = bid % (MB64 * 32);
    int mB    = rem / 32;
    int nT    = rem % 32;
    int kHalf = lane >> 4;
    int l15   = lane & 15;
    int mBase = mB * 64 + wave * 16;
    int nBase = nT * 64;

    int mRow = mBase + l15; if (mRow > LE - 1) mRow = LE - 1;
    const __bf16* aBase = frames + ((size_t)bc * LE + mRow) * NFFT + kHalf * 16;

    // --- per-lane async staging plan: 4 x 16B chunks per thread per k-step ---
    unsigned ldsBase = (unsigned)(uintptr_t)(lds_bf16*)&smem[0];
    const __bf16* gRow[4];
    unsigned ldsOff[4];
    #pragma unroll
    for (int i = 0; i < 4; ++i) {
        int basis = i >> 1;                       // 0=cos 1=sin
        int r     = (((i & 1) << 7) + lid) >> 2;  // 0..63 (row within tile)
        int col   = lid & 3;                      // 16B chunk within 64B row
        const __bf16* gb = basis ? sinT : cosT;
        gRow[i]   = gb + (size_t)(nBase + r) * NFFT + col * 8;
        ldsOff[i] = (unsigned)((basis * BASIS_BF + r * ROW_BF + col * 8) * 2);
    }

    // per-wave fragment offsets within one basis tile
    int fro[4];
    #pragma unroll
    for (int s = 0; s < 4; ++s) fro[s] = (s * 16 + l15) * ROW_BF + kHalf * 16;

    v8f zero = {0.f,0.f,0.f,0.f,0.f,0.f,0.f,0.f};
    v8f accR[4] = {zero, zero, zero, zero};
    v8f accS[4] = {zero, zero, zero, zero};

    constexpr int STEPS = NFFT / 32;   // 128

    // prologue: stage k=0 into buffer 0
    #pragma unroll
    for (int i = 0; i < 4; ++i) async_ld16(ldsBase + ldsOff[i], gRow[i]);
    wait_async0();
    __syncthreads();

    v16bf aCur = load_a16(aBase);

    for (int ks = 0; ks < STEPS; ++ks) {
        int pb    = ks & 1;
        int knext = (ks + 1 < STEPS) ? (ks + 1) * 32 : (NFFT - 32);
        // issue async stage of next k-step into the other buffer
        unsigned pbase = ldsBase + (unsigned)((pb ^ 1) * PB_BF * 2);
        #pragma unroll
        for (int i = 0; i < 4; ++i) async_ld16(pbase + ldsOff[i], gRow[i] + knext);
        // prefetch next A fragment
        v16bf aNxt = load_a16(aBase + knext);

        // burst-load all 8 B fragments from current buffer, then dense WMMA run
        const __bf16* smC = smem + pb * PB_BF;
        const __bf16* smS = smC + BASIS_BF;
        v16bf bC[4], bS[4];
        #pragma unroll
        for (int s = 0; s < 4; ++s) {
            bC[s] = *(const v16bf*)(smC + fro[s]);
            bS[s] = *(const v16bf*)(smS + fro[s]);
        }
        #pragma unroll
        for (int s = 0; s < 4; ++s) {
            accR[s] = wmma_bf16(aCur, bC[s], accR[s]);
            accS[s] = wmma_bf16(aCur, bS[s], accS[s]);
        }
        aCur = aNxt;
        wait_async0();
        __syncthreads();
    }

    #pragma unroll
    for (int s = 0; s < 4; ++s) {
        #pragma unroll
        for (int g = 0; g < 8; ++g) {
            int m = mBase + g + kHalf * 8;
            if (m < LE) {
                int n = nBase + s * 16 + l15;
                size_t o = ((size_t)bc * LE + m) * NF + n;
                Zr[o] = accR[s][g];
                Si[o] = accS[s][g];
            }
        }
    }
}

// ---------------------------------------------------------------------------
// K3: mask multiply + bf16 convert + transpose mask [f][le] -> [le][f] via LDS.
// A2r =  Zr*maskR ; A2i = -(Si*maskI)  (signs set so iSTFT is Ar*cos + Ai*sin)
// ---------------------------------------------------------------------------
__global__ __launch_bounds__(256) void mask_mul(
    const float* __restrict__ Zr, const float* __restrict__ Si,
    const float* __restrict__ mask,
    __bf16* __restrict__ A2r, __bf16* __restrict__ A2i)
{
    __shared__ float lr[16][17];
    __shared__ float li[16][17];
    int tx = threadIdx.x, ty = threadIdx.y;
    int fBase  = blockIdx.x * 16;
    int leBase = blockIdx.y * 16;
    int z = blockIdx.z;                 // bsc = b*8 + s*2 + c
    int b = z >> 3, s = (z >> 1) & 3, c = z & 1;
    int bc = b * CC + c;

    {   // coalesced mask read (contiguous in le)
        int le = leBase + tx, f = fBase + ty;
        float mr = 0.f, mi = 0.f;
        if (le < LE) {
            size_t base = ((((size_t)b * SS + s) * (2 * CC) + c) * NF + f) * LE + le;
            mr = mask[base];
            mi = mask[base + (size_t)CC * NF * LE];
        }
        lr[ty][tx] = mr;
        li[ty][tx] = mi;
    }
    __syncthreads();
    {   // coalesced Z read / A2 write (contiguous in f)
        int le = leBase + ty, f = fBase + tx;
        if (le < LE) {
            size_t zo = ((size_t)bc * LE + le) * NF + f;
            size_t ao = ((size_t)z  * LE + le) * NF + f;
            A2r[ao] = (__bf16)(Zr[zo] * lr[tx][ty]);
            A2i[ao] = (__bf16)(-(Si[zo] * li[tx][ty]));
        }
    }
}

// ---------------------------------------------------------------------------
// K4: iSTFT GEMM.  M=le(431) N=t(4096) K=f(2048).
// FS[bsc][le][t] = sum_f A2r*cos + A2i*sin  (pre-window frame signal, bf16)
// Same async-LDS double-buffered staging as K2, burst B-fragment loads.
// ---------------------------------------------------------------------------
__global__ __launch_bounds__(128) void gemm_istft(
    const __bf16* __restrict__ A2r, const __bf16* __restrict__ A2i,
    const __bf16* __restrict__ cosTT, const __bf16* __restrict__ sinTT,
    __bf16* __restrict__ FS)
{
    __shared__ alignas(16) __bf16 smem[SMEM_BF];

    int lid   = threadIdx.x;
    int lane  = lid & 31;
    int wave  = lid >> 5;
    int bid   = blockIdx.x;            // < 32 * 7 * 64
    int bsc   = bid / (MB64 * 64);
    int rem   = bid % (MB64 * 64);
    int mB    = rem / 64;
    int nT    = rem % 64;
    int kHalf = lane >> 4;
    int l15   = lane & 15;
    int mBase = mB * 64 + wave * 16;
    int nBase = nT * 64;

    int mRow = mBase + l15; if (mRow > LE - 1) mRow = LE - 1;
    const __bf16* aRb = A2r + ((size_t)bsc * LE + mRow) * NF + kHalf * 16;
    const __bf16* aIb = A2i + ((size_t)bsc * LE + mRow) * NF + kHalf * 16;

    unsigned ldsBase = (unsigned)(uintptr_t)(lds_bf16*)&smem[0];
    const __bf16* gRow[4];
    unsigned ldsOff[4];
    #pragma unroll
    for (int i = 0; i < 4; ++i) {
        int basis = i >> 1;
        int r     = (((i & 1) << 7) + lid) >> 2;
        int col   = lid & 3;
        const __bf16* gb = basis ? sinTT : cosTT;
        gRow[i]   = gb + (size_t)(nBase + r) * NF + col * 8;
        ldsOff[i] = (unsigned)((basis * BASIS_BF + r * ROW_BF + col * 8) * 2);
    }

    int fro[4];
    #pragma unroll
    for (int s = 0; s < 4; ++s) fro[s] = (s * 16 + l15) * ROW_BF + kHalf * 16;

    v8f zero = {0.f,0.f,0.f,0.f,0.f,0.f,0.f,0.f};
    v8f acc[4] = {zero, zero, zero, zero};

    constexpr int STEPS = NF / 32;     // 64

    #pragma unroll
    for (int i = 0; i < 4; ++i) async_ld16(ldsBase + ldsOff[i], gRow[i]);
    wait_async0();
    __syncthreads();

    v16bf aRc = load_a16(aRb);
    v16bf aIc = load_a16(aIb);

    for (int ks = 0; ks < STEPS; ++ks) {
        int pb    = ks & 1;
        int knext = (ks + 1 < STEPS) ? (ks + 1) * 32 : (NF - 32);
        unsigned pbase = ldsBase + (unsigned)((pb ^ 1) * PB_BF * 2);
        #pragma unroll
        for (int i = 0; i < 4; ++i) async_ld16(pbase + ldsOff[i], gRow[i] + knext);
        v16bf aRn = load_a16(aRb + knext);
        v16bf aIn = load_a16(aIb + knext);

        const __bf16* smC = smem + pb * PB_BF;
        const __bf16* smS = smC + BASIS_BF;
        v16bf bC[4], bS[4];
        #pragma unroll
        for (int s = 0; s < 4; ++s) {
            bC[s] = *(const v16bf*)(smC + fro[s]);
            bS[s] = *(const v16bf*)(smS + fro[s]);
        }
        #pragma unroll
        for (int s = 0; s < 4; ++s) {
            acc[s] = wmma_bf16(aRc, bC[s], acc[s]);
            acc[s] = wmma_bf16(aIc, bS[s], acc[s]);
        }
        aRc = aRn;
        aIc = aIn;
        wait_async0();
        __syncthreads();
    }

    #pragma unroll
    for (int s = 0; s < 4; ++s) {
        #pragma unroll
        for (int g = 0; g < 8; ++g) {
            int m = mBase + g + kHalf * 8;
            if (m < LE) {
                int n = nBase + s * 16 + l15;
                FS[((size_t)bsc * LE + m) * NFFT + n] = (__bf16)acc[s][g];
            }
        }
    }
}

// ---------------------------------------------------------------------------
// K5: windowed overlap-add. norm == 1.5 exactly for the output range.
// ---------------------------------------------------------------------------
__global__ __launch_bounds__(256) void overlap_out(
    const __bf16* __restrict__ FS, float* __restrict__ out)
{
    int idx = blockIdx.x * 256 + threadIdx.x;          // < 32*441000
    if (idx >= NBSC * TLEN) return;
    int j   = idx % TLEN;
    int bsc = idx / TLEN;
    int p   = j + 3584;
    int o   = p & 1023;
    int nhi = p >> 10;
    float acc = 0.f;
    #pragma unroll
    for (int k = 0; k < 4; ++k) {
        int le = nhi - k - 2;
        if (le >= 0 && le < LE) {
            int t = o + (k << 10);
            float w = 0.5f - 0.5f * __cosf((float)t * TWO_PI_OVER_N);
            acc += w * (float)FS[((size_t)bsc * LE + le) * NFFT + t];
        }
    }
    out[idx] = acc * (2.0f / 3.0f);
}

// ---------------------------------------------------------------------------
extern "C" void kernel_launch(void* const* d_in, const int* in_sizes, int n_in,
                              void* d_out, int out_size, void* d_ws, size_t ws_size,
                              hipStream_t stream) {
    const float* mix  = (const float*)d_in[0];   // [4,2,441000]
    const float* mask = (const float*)d_in[1];   // [4,4,4,2048,431]
    float* out = (float*)d_out;                  // [4,4,2,441000]

    // ---- workspace bump allocator ----
    char* ws = (char*)d_ws;
    size_t off = 0;
    auto alloc = [&](size_t bytes) -> void* {
        void* p = ws + off;
        off = (off + bytes + 255) & ~(size_t)255;
        return p;
    };
    __bf16* frames = (__bf16*)alloc((size_t)NBC * LE * NFFT * 2);     // 28.2 MB
    __bf16* cosT   = (__bf16*)alloc((size_t)NF * NFFT * 2);           // 16.8 MB
    __bf16* sinT   = (__bf16*)alloc((size_t)NF * NFFT * 2);
    __bf16* cosTT  = (__bf16*)alloc((size_t)NFFT * NF * 2);
    __bf16* sinTT  = (__bf16*)alloc((size_t)NFFT * NF * 2);
    float*  Zr     = (float*) alloc((size_t)NBC * LE * NF * 4);       // 28.2 MB
    float*  Si     = (float*) alloc((size_t)NBC * LE * NF * 4);
    __bf16* A2r    = (__bf16*)alloc((size_t)NBSC * LE * NF * 2);      // 56.5 MB
    __bf16* A2i    = (__bf16*)alloc((size_t)NBSC * LE * NF * 2);
    __bf16* FS     = (__bf16*)alloc((size_t)NBSC * LE * NFFT * 2);    // 113  MB
    if (off > ws_size) return;  // workspace too small -> refuse to scribble

    build_tables<<<(NF * NFFT) / 256, 256, 0, stream>>>(cosT, sinT, cosTT, sinTT);
    build_frames<<<(NBC * LE * NFFT) / 256, 256, 0, stream>>>(mix, frames);

    // STFT GEMM: 8 * 7 * 32 blocks of 4 waves (64x64 tile each)
    gemm_stft<<<NBC * MB64 * 32, 128, 0, stream>>>(frames, cosT, sinT, Zr, Si);

    mask_mul<<<dim3(NF / 16, 27, NBSC), dim3(16, 16), 0, stream>>>(Zr, Si, mask, A2r, A2i);

    // iSTFT GEMM: 32 * 7 * 64 blocks of 4 waves
    gemm_istft<<<NBSC * MB64 * 64, 128, 0, stream>>>(A2r, A2i, cosTT, sinTT, FS);

    overlap_out<<<(NBSC * TLEN + 255) / 256, 256, 0, stream>>>(FS, out);
}
